// WideDeep_43095701848227
// MI455X (gfx1250) — matrix operand order repaired and checked by reading, not compiled, not run on hardware
//
#include <hip/hip_runtime.h>
#include <cstdint>
#include <cstddef>

typedef __bf16 v16bf __attribute__((ext_vector_type(16)));
typedef float  v8f   __attribute__((ext_vector_type(8)));
typedef unsigned short u16;

#define BATCH 8192
#define KPAD1 768   // 748 padded to multiple of 32

__device__ __forceinline__ u16 f2bf(float f) {
  union { float f; unsigned u; } x; x.f = f;
  unsigned r = x.u + 0x7fffu + ((x.u >> 16) & 1u);   // round-to-nearest-even
  return (u16)(r >> 16);
}
__device__ __forceinline__ float bf2f(u16 h) {
  union { unsigned u; float f; } x; x.u = ((unsigned)h) << 16;
  return x.f;
}

// ---------------------------------------------------------------------------
// Stage 1: embedding pooling -> bf16 activation row [768], plus wide scalar.
// One block (256 threads) per sample.
// ---------------------------------------------------------------------------
__global__ __launch_bounds__(256) void embed_kernel(
    const int* __restrict__ tok, const int* __restrict__ mh1,
    const int* __restrict__ mh2, const int* __restrict__ mh3,
    const int* __restrict__ oh1, const int* __restrict__ oh2,
    const int* __restrict__ oh3, const int* __restrict__ oh4,
    const float* __restrict__ we,
    const float* __restrict__ e1, const float* __restrict__ e2,
    const float* __restrict__ e3,
    const float* __restrict__ o1, const float* __restrict__ o2,
    const float* __restrict__ o3, const float* __restrict__ o4,
    const float* __restrict__ wm1, const float* __restrict__ wm2,
    const float* __restrict__ wm3,
    const float* __restrict__ wo1, const float* __restrict__ wo2,
    const float* __restrict__ wo3, const float* __restrict__ wo4,
    u16* __restrict__ embB, float* __restrict__ wide) {
  __shared__ int s_tok[32];
  __shared__ int s_mh0[20], s_mh1[20], s_mh2[20];
  __shared__ float s_red[64];

  const int b   = blockIdx.x;
  const int tid = threadIdx.x;

  if (tid < 32)              s_tok[tid]       = tok[b * 32 + tid];
  else if (tid < 52)         s_mh0[tid - 32]  = mh1[b * 20 + (tid - 32)];
  else if (tid < 72)         s_mh1[tid - 52]  = mh2[b * 20 + (tid - 52)];
  else if (tid < 92)         s_mh2[tid - 72]  = mh3[b * 20 + (tid - 72)];
  __syncthreads();

  u16* row = embB + (size_t)b * KPAD1;

  // token EmbeddingBag(mean) over 32 tokens, dim 300
  for (int d = tid; d < 300; d += 256) {
    float s = 0.0f;
    #pragma unroll 4
    for (int t = 0; t < 32; ++t) s += we[(size_t)s_tok[t] * 300 + d];
    row[d] = f2bf(s * (1.0f / 32.0f));
  }

  // three multi-hot bags (mean over 20), dim 64 each -> cols [300, 492)
  for (int j = tid; j < 192; j += 256) {
    const int f = j >> 6, d = j & 63;
    const int*   mi = (f == 0) ? s_mh0 : (f == 1) ? s_mh1 : s_mh2;
    const float* E  = (f == 0) ? e1    : (f == 1) ? e2    : e3;
    float s = 0.0f;
    #pragma unroll 4
    for (int t = 0; t < 20; ++t) s += E[(size_t)mi[t] * 64 + d];
    row[300 + j] = f2bf(s * (1.0f / 20.0f));
  }

  // four one-hot embeddings, dim 64 each -> cols [492, 748)
  {
    const int j = tid;                       // 0..255
    const int f = j >> 6, d = j & 63;
    const int    ix = (f == 0) ? oh1[b] : (f == 1) ? oh2[b] : (f == 2) ? oh3[b] : oh4[b];
    const float* O  = (f == 0) ? o1     : (f == 1) ? o2     : (f == 2) ? o3     : o4;
    row[492 + j] = f2bf(O[(size_t)ix * 64 + d]);
  }

  // zero pad cols [748, 768)
  if (tid < 20) row[748 + tid] = 0;

  // wide tower: 60 multi-hot + 4 one-hot 1-dim values summed
  float wv = 0.0f;
  if (tid < 60) {
    const int f = tid / 20, t = tid % 20;
    wv = (f == 0) ? wm1[s_mh0[t]] : (f == 1) ? wm2[s_mh1[t]] : wm3[s_mh2[t]];
  } else if (tid < 64) {
    const int f = tid - 60;
    wv = (f == 0) ? wo1[oh1[b]] : (f == 1) ? wo2[oh2[b]]
       : (f == 2) ? wo3[oh3[b]] : wo4[oh4[b]];
  }
  if (tid < 64) s_red[tid] = wv;
  __syncthreads();
  if (tid == 0) {
    float s = 0.0f;
    #pragma unroll
    for (int i = 0; i < 64; ++i) s += s_red[i];
    wide[b] = s;
  }
}

// ---------------------------------------------------------------------------
// Stage 2: convert f32 weights [K,N] row-major -> bf16 packed in this GEMM's
// WMMA B-fragment layout (per 32x16 K-N tile: lane l = (K-half<<4)|n, 16
// contiguous bf16 per lane covering K offsets 16*(l>>4)+e). Zero-pads K->Kp.
// ---------------------------------------------------------------------------
__global__ __launch_bounds__(256) void pack_w_kernel(
    const float* __restrict__ W, u16* __restrict__ out, int K, int N, int Kp) {
  const int idx   = blockIdx.x * 256 + threadIdx.x;
  const int total = Kp * N;                 // packed element count
  if (idx >= total) return;
  const int Kt   = Kp >> 5;
  const int e    = idx & 15;
  const int lane = (idx >> 4) & 31;
  const int t    = idx >> 9;                // tile id = nt*Kt + kt
  const int kt   = t % Kt;
  const int nt   = t / Kt;
  const int n = nt * 16 + (lane & 15);
  const int k = kt * 32 + ((lane >> 4) << 4) + e;
  const float v = (k < K) ? W[(size_t)k * N + n] : 0.0f;
  out[idx] = f2bf(v);
}

// ---------------------------------------------------------------------------
// Stage 3: GEMM  C[M,N] = relu(A[M,Kp] @ W + bias), bf16 in / f32 acc / bf16
// out on v_wmma_f32_16x16x32_bf16. One wave computes a 32x64 output tile
// (2 m-tiles x 4 n-tiles, 8 accumulators): per 32-K step the wave issues
// 8 WMMAs against 12 b128 loads (4 A + 8 B), with A fragments reused across
// the 4 N-tiles and each B fragment reused across the 2 M-tiles.
// Grid: (M/256, N/64), 256 threads = 8 waves.
// ---------------------------------------------------------------------------
__global__ __launch_bounds__(256) void gemm_bf16_wmma(
    const u16* __restrict__ A, const u16* __restrict__ Bp,
    const float* __restrict__ bias, u16* __restrict__ C, int N, int Kp) {
  const int lane = threadIdx.x & 31;
  const int wid  = threadIdx.x >> 5;
  const int mt0  = (blockIdx.x * 8 + wid) * 2;  // first of 2 16-row tiles
  const int ng   = blockIdx.y;                  // group of 4 16-col tiles
  const int Kt   = Kp >> 5;
  const int h    = lane >> 4;                   // K-half select
  const int ml   = lane & 15;                   // A row within tile

  v8f acc[2][4] = {};

  // A k order used here (matches B packing): lane (h,ml) holds K = {16h..16h+15}
  const u16* arow0 = A + (size_t)(mt0 * 16 + ml) * Kp + 16 * h;
  const u16* arow1 = arow0 + (size_t)16 * Kp;

  for (int kt = 0; kt < Kt; ++kt) {
    v16bf a0, a1;
    const u16* ap0 = arow0 + kt * 32;
    const u16* ap1 = arow1 + kt * 32;
    ((uint4*)&a0)[0] = *(const uint4*)(ap0);
    ((uint4*)&a0)[1] = *(const uint4*)(ap0 + 8);
    ((uint4*)&a1)[0] = *(const uint4*)(ap1);
    ((uint4*)&a1)[1] = *(const uint4*)(ap1 + 8);
    #pragma unroll
    for (int i = 0; i < 4; ++i) {
      const int nt = (ng << 2) + i;
      const u16* bp = Bp + ((((size_t)nt * Kt + kt) * 32 + lane) << 4);
      v16bf bfr;
      ((uint4*)&bfr)[0] = ((const uint4*)bp)[0];
      ((uint4*)&bfr)[1] = ((const uint4*)bp)[1];
      acc[0][i] = __builtin_amdgcn_wmma_f32_16x16x32_bf16(
          false, a0, false, bfr, (short)0, acc[0][i], false, false);
      acc[1][i] = __builtin_amdgcn_wmma_f32_16x16x32_bf16(
          false, a1, false, bfr, (short)0, acc[1][i], false, false);
    }
  }

  // C layout: VGPR r <-> M = 8*h + r, N = ml
  #pragma unroll
  for (int mi = 0; mi < 2; ++mi) {
    #pragma unroll
    for (int i = 0; i < 4; ++i) {
      const int n  = ((ng << 2) + i) * 16 + ml;
      const float bv = bias[n];
      #pragma unroll
      for (int r = 0; r < 8; ++r) {
        const int m = (mt0 + mi) * 16 + h * 8 + r;
        float v = acc[mi][i][r] + bv;
        v = v > 0.0f ? v : 0.0f;
        C[(size_t)m * N + n] = f2bf(v);
      }
    }
  }
}

// ---------------------------------------------------------------------------
// Stage 4: head  out[b] = sigmoid(h3[b,:] . Wd + bd + wide[b]).
// One wave per sample, shuffle reduction (wave32).
// ---------------------------------------------------------------------------
__global__ __launch_bounds__(256) void dense_out_kernel(
    const u16* __restrict__ h3, const float* __restrict__ Wd,
    const float* __restrict__ bd, const float* __restrict__ wide,
    float* __restrict__ out) {
  const int lane = threadIdx.x & 31;
  const int wid  = threadIdx.x >> 5;
  const int s    = blockIdx.x * 8 + wid;
  const u16* hr = h3 + (size_t)s * 256;
  float sum = 0.0f;
  #pragma unroll
  for (int i = 0; i < 8; ++i) {
    const int d = lane + 32 * i;
    sum += bf2f(hr[d]) * Wd[d];
  }
  #pragma unroll
  for (int off = 16; off > 0; off >>= 1) sum += __shfl_down(sum, off, 32);
  if (lane == 0) {
    const float x = sum + bd[0] + wide[s];
    out[s] = 1.0f / (1.0f + __expf(-x));
  }
}

// ---------------------------------------------------------------------------
extern "C" void kernel_launch(void* const* d_in, const int* in_sizes, int n_in,
                              void* d_out, int out_size, void* d_ws, size_t ws_size,
                              hipStream_t stream) {
  const int*   tok = (const int*)d_in[0];
  const int*   mh1 = (const int*)d_in[1];
  const int*   mh2 = (const int*)d_in[2];
  const int*   mh3 = (const int*)d_in[3];
  const int*   oh1 = (const int*)d_in[4];
  const int*   oh2 = (const int*)d_in[5];
  const int*   oh3 = (const int*)d_in[6];
  const int*   oh4 = (const int*)d_in[7];
  const float* we  = (const float*)d_in[8];
  const float* e1  = (const float*)d_in[9];
  const float* e2  = (const float*)d_in[10];
  const float* e3  = (const float*)d_in[11];
  const float* o1  = (const float*)d_in[12];
  const float* o2  = (const float*)d_in[13];
  const float* o3  = (const float*)d_in[14];
  const float* o4  = (const float*)d_in[15];
  const float* wm1 = (const float*)d_in[16];
  const float* wm2 = (const float*)d_in[17];
  const float* wm3 = (const float*)d_in[18];
  const float* wo1 = (const float*)d_in[19];
  const float* wo2 = (const float*)d_in[20];
  const float* wo3 = (const float*)d_in[21];
  const float* wo4 = (const float*)d_in[22];
  const float* W1  = (const float*)d_in[23];
  const float* b1  = (const float*)d_in[24];
  const float* W2  = (const float*)d_in[25];
  const float* b2  = (const float*)d_in[26];
  const float* W3  = (const float*)d_in[27];
  const float* b3  = (const float*)d_in[28];
  const float* Wd  = (const float*)d_in[29];
  const float* bd  = (const float*)d_in[30];
  float* out = (float*)d_out;

  // Workspace carve-up (all sizes 256B-aligned by construction)
  char* p = (char*)d_ws;
  size_t o = 0;
  u16* embB = (u16*)(p + o); o += (size_t)BATCH * KPAD1 * 2;   // 12.6 MB
  u16* W1p  = (u16*)(p + o); o += (size_t)KPAD1 * 1024 * 2;    //  1.6 MB
  u16* h1   = (u16*)(p + o); o += (size_t)BATCH * 1024 * 2;    // 16.8 MB
  u16* W2p  = (u16*)(p + o); o += (size_t)1024 * 512 * 2;      //  1.0 MB
  u16* h2   = (u16*)(p + o); o += (size_t)BATCH * 512 * 2;     //  8.4 MB
  u16* W3p  = (u16*)(p + o); o += (size_t)512 * 256 * 2;       //  0.3 MB
  u16* h3   = (u16*)(p + o); o += (size_t)BATCH * 256 * 2;     //  4.2 MB
  float* wide = (float*)(p + o); o += (size_t)BATCH * 4;       // 32 KB

  // 1) embedding pooling + wide tower
  embed_kernel<<<BATCH, 256, 0, stream>>>(
      tok, mh1, mh2, mh3, oh1, oh2, oh3, oh4,
      we, e1, e2, e3, o1, o2, o3, o4,
      wm1, wm2, wm3, wo1, wo2, wo3, wo4, embB, wide);

  // 2) weight conversion + B-fragment packing
  {
    int t1 = KPAD1 * 1024;
    pack_w_kernel<<<(t1 + 255) / 256, 256, 0, stream>>>(W1, W1p, 748, 1024, KPAD1);
    int t2 = 1024 * 512;
    pack_w_kernel<<<(t2 + 255) / 256, 256, 0, stream>>>(W2, W2p, 1024, 512, 1024);
    int t3 = 512 * 256;
    pack_w_kernel<<<(t3 + 255) / 256, 256, 0, stream>>>(W3, W3p, 512, 256, 512);
  }

  // 3) three WMMA GEMM layers with fused bias+ReLU (32x64 tile per wave)
  gemm_bf16_wmma<<<dim3(BATCH / 256, 1024 / 64), 256, 0, stream>>>(embB, W1p, b1, h1, 1024, KPAD1);
  gemm_bf16_wmma<<<dim3(BATCH / 256,  512 / 64), 256, 0, stream>>>(h1,   W2p, b2, h2,  512, 1024);
  gemm_bf16_wmma<<<dim3(BATCH / 256,  256 / 64), 256, 0, stream>>>(h2,   W3p, b3, h3,  256,  512);

  // 4) dense head + wide + sigmoid
  dense_out_kernel<<<BATCH / 8, 256, 0, stream>>>(h3, Wd, bd, wide, out);
}
